// GRUModel_3246995276512
// MI455X (gfx1250) — matrix operand order repaired and checked by reading, not compiled, not run on hardware
//
#include <hip/hip_runtime.h>

typedef _Float16 v16h __attribute__((ext_vector_type(16)));
typedef _Float16 v8h  __attribute__((ext_vector_type(8)));
typedef __fp16   v2hp __attribute__((ext_vector_type(2)));   // __builtin_amdgcn_cvt_pkrtz return type
typedef float    v8f  __attribute__((ext_vector_type(8)));

#define BATCH 8192
#define TT    512
#define FF    8
#define HH    32
#define GG    96      // 3*H gates
#define WAVES 4
#define BLOCK (WAVES * 32)
#define BPW   16      // batch rows per wave (WMMA M)
#define LOG2E 1.4426950408889634f

__device__ __forceinline__ float fast_sigmoid(float x) {
    return __builtin_amdgcn_rcpf(1.f + __builtin_amdgcn_exp2f(-LOG2E * x));
}
__device__ __forceinline__ float fast_tanh(float x) {
    // tanh(x) = 2*sigmoid(2x) - 1 ; saturates correctly via exp2 overflow/underflow
    float e = __builtin_amdgcn_exp2f(-2.f * LOG2E * x);
    return fmaf(2.f, __builtin_amdgcn_rcpf(1.f + e), -1.f);
}

__launch_bounds__(BLOCK, 1)
__global__ void gru_fused_kernel(const float* __restrict__ x,
                                 const float* __restrict__ W_ih,
                                 const float* __restrict__ W_hh,
                                 const float* __restrict__ b_ih,
                                 const float* __restrict__ b_hh,
                                 const float* __restrict__ W_fc,
                                 const float* __restrict__ b_fc,
                                 float* __restrict__ out)
{
    // Combined f16 weight rows, K-permuted hidden part:
    //   K=2u   -> W_hh[g][u],  K=2u+1 -> W_hh[g][16+u]   (u = 0..15)
    //   K'=0..7 (halves 32..39) -> W_ih[g][0..7]
    //   K'=8    (half 40)       -> bias rider (b_ih+b_hh for r/z gates, b_ih for n gates)
    __shared__ __align__(32) _Float16 wlds[GG][64];
    // Per-wave hidden state staging (f16), padded stride 40
    __shared__ __align__(16) _Float16 hlds[WAVES][BPW][40];
    __shared__ float wfc_s[HH];
    __shared__ float fout[WAVES][BPW][HH + 1];

    const int tid  = threadIdx.x;
    const int w    = tid >> 5;
    const int lane = tid & 31;
    const int hf   = lane >> 4;
    const int ln   = lane & 15;
    const int bb   = blockIdx.x * (WAVES * BPW) + w * BPW;

    // ---- stage weights (f32 -> f16) into LDS, zero h state ----
    if (tid < GG) {
#pragma unroll
        for (int u = 0; u < 16; ++u) {
            wlds[tid][2 * u]     = (_Float16)W_hh[tid * HH + u];
            wlds[tid][2 * u + 1] = (_Float16)W_hh[tid * HH + 16 + u];
        }
#pragma unroll
        for (int f = 0; f < FF; ++f) wlds[tid][HH + f] = (_Float16)W_ih[tid * FF + f];
        const float bias = (tid < 2 * HH) ? (b_ih[tid] + b_hh[tid]) : b_ih[tid];
        wlds[tid][40] = (_Float16)bias;
#pragma unroll
        for (int k = 41; k < 64; ++k) wlds[tid][k] = (_Float16)0.f;
    }
    if (tid < HH) wfc_s[tid] = W_fc[tid];
    for (int i = tid; i < WAVES * BPW * 40; i += BLOCK)
        ((_Float16*)hlds)[i] = (_Float16)0.f;
    __syncthreads();

    // ---- preload B-matrix weight tiles into registers (12 tiles x 8 VGPRs) ----
    // B layout (16-bit 32x16): lane column g = nt*16+ln; 16 contiguous K-halves at K = hf*16.
    v16h Blo[6], Bhi[6];
#pragma unroll
    for (int nt = 0; nt < 6; ++nt) {
        const int g = nt * 16 + ln;
        Blo[nt] = *(const v16h*)&wlds[g][hf * 16];        // hidden K-chunk
        Bhi[nt] = *(const v16h*)&wlds[g][32 + hf * 16];   // input K-chunk (+ bias rider row)
    }
    // only n-gate hidden bias is applied elementwise (it sits inside r*(h_n + b))
    float bHn[2];
#pragma unroll
    for (int j = 0; j < 2; ++j) bHn[j] = b_hh[2 * HH + j * 16 + ln];

    // hidden state in C/D layout: h[j][v] = h[batch v+8*hf][unit j*16+ln]
    v8f h[2];
#pragma unroll
    for (int j = 0; j < 2; ++j)
#pragma unroll
        for (int v = 0; v < 8; ++v) h[j][v] = 0.f;

    const int    k0   = hf * 16;
    const float* xrow = x + (size_t)(bb + ln) * TT * FF;  // all lanes load (16..31 redundant)
    const float  xm   = (hf == 0) ? 1.f : 0.f;            // lane-half mask, branchless
    const float  xc1  = 1.f - xm;                         // complement of xm (unused; K'=8 rider uses xm)

    float xr[8];
    {
        const float4* p = (const float4*)xrow;
        float4 a = p[0], b = p[1];
        xr[0] = a.x; xr[1] = a.y; xr[2] = a.z; xr[3] = a.w;
        xr[4] = b.x; xr[5] = b.y; xr[6] = b.z; xr[7] = b.w;
    }

    for (int t = 0; t < TT; ++t) {
        // A_lo (16x32 f16, hidden): lane<16 -> K 0..15; lane>=16 -> K 16..31
        v8h p0 = *(const v8h*)&hlds[w][ln][k0];
        v8h p1 = *(const v8h*)&hlds[w][ln][k0 + 8];
        v16h Alo = __builtin_shufflevector(p0, p1, 0,1,2,3,4,5,6,7,8,9,10,11,12,13,14,15);

        // A_hi: lanes<16 -> K'0..7 = x_t, K'8 = 1.0 (bias rider); lanes>=16 -> zero
        v16h Ahi = {};
        Ahi[8] = (_Float16)xm;
#pragma unroll
        for (int f = 0; f < FF; ++f) Ahi[f] = (_Float16)(xr[f] * xm);

        // branchless prefetch of next timestep's x
        {
            const int tn = (t + 1 < TT) ? (t + 1) : (TT - 1);
            const float4* p = (const float4*)(xrow + (size_t)tn * FF);
            float4 a = p[0], b = p[1];
            xr[0] = a.x; xr[1] = a.y; xr[2] = a.z; xr[3] = a.w;
            xr[4] = b.x; xr[5] = b.y; xr[6] = b.z; xr[7] = b.w;
        }

        const v8f zero8 = {};   // folds to inline SRC2=0
#pragma unroll
        for (int j = 0; j < 2; ++j) {
            v8f aR = __builtin_amdgcn_wmma_f32_16x16x32_f16(false, Alo, false, Blo[j],     (short)0, zero8, false, false);
            v8f pR = __builtin_amdgcn_wmma_f32_16x16x32_f16(false, Ahi, false, Bhi[j],     (short)0, aR,    false, false);
            v8f aZ = __builtin_amdgcn_wmma_f32_16x16x32_f16(false, Alo, false, Blo[2 + j], (short)0, zero8, false, false);
            v8f pZ = __builtin_amdgcn_wmma_f32_16x16x32_f16(false, Ahi, false, Bhi[2 + j], (short)0, aZ,    false, false);
            v8f hn = __builtin_amdgcn_wmma_f32_16x16x32_f16(false, Alo, false, Blo[4 + j], (short)0, zero8, false, false);
            v8f xi = __builtin_amdgcn_wmma_f32_16x16x32_f16(false, Ahi, false, Bhi[4 + j], (short)0, zero8, false, false);
#pragma unroll
            for (int v = 0; v < 8; ++v) {
                const float r  = fast_sigmoid(pR[v]);
                const float z  = fast_sigmoid(pZ[v]);
                const float nv = fast_tanh(fmaf(r, hn[v] + bHn[j], xi[v]));
                h[j][v] = nv + z * (h[j][v] - nv);        // (1-z)*n + z*h
            }
        }

        // writeback h: units (ln, 16+ln) pack into one dword at K halves (2ln, 2ln+1)
#pragma unroll
        for (int v = 0; v < 8; ++v) {
            v2hp pk = __builtin_amdgcn_cvt_pkrtz(h[0][v], h[1][v]);
            *(v2hp*)&hlds[w][v + 8 * hf][2 * ln] = pk;
        }
        __asm__ volatile("s_wait_dscnt 0" ::: "memory");  // wave-private region, DS-order only
    }

    // ---- epilogue: out[b] = h_last . W_fc + b_fc ----
#pragma unroll
    for (int j = 0; j < 2; ++j)
#pragma unroll
        for (int v = 0; v < 8; ++v)
            fout[w][v + 8 * hf][j * 16 + ln] = h[j][v];
    __asm__ volatile("s_wait_dscnt 0" ::: "memory");

    if (hf == 0) {
        float acc = b_fc[0];
#pragma unroll
        for (int u = 0; u < HH; ++u) acc += fout[w][ln][u] * wfc_s[u];
        out[bb + ln] = acc;
    }
}

extern "C" void kernel_launch(void* const* d_in, const int* in_sizes, int n_in,
                              void* d_out, int out_size, void* d_ws, size_t ws_size,
                              hipStream_t stream) {
    (void)in_sizes; (void)n_in; (void)d_ws; (void)ws_size; (void)out_size;
    const float* x    = (const float*)d_in[0];
    const float* W_ih = (const float*)d_in[1];
    const float* W_hh = (const float*)d_in[2];
    const float* b_ih = (const float*)d_in[3];
    const float* b_hh = (const float*)d_in[4];
    const float* W_fc = (const float*)d_in[5];
    const float* b_fc = (const float*)d_in[6];
    float* out = (float*)d_out;

    dim3 grid(BATCH / (WAVES * BPW));   // 128 blocks
    dim3 block(BLOCK);                  // 4 waves of 32
    gru_fused_kernel<<<grid, block, 0, stream>>>(x, W_ih, W_hh, b_ih, b_hh, W_fc, b_fc, out);
}
